// get_model_1451698946610
// MI455X (gfx1250) — matrix operand order repaired and checked by reading, not compiled, not run on hardware
//
#include <hip/hip_runtime.h>
#include <hip/hip_bf16.h>
#include <math.h>

// ---------------------------------------------------------------------------
// Types for CDNA5 WMMA
// ---------------------------------------------------------------------------
typedef __attribute__((ext_vector_type(16))) __bf16          v16bf;
typedef __attribute__((ext_vector_type(16))) unsigned short  v16us;
typedef __attribute__((ext_vector_type(8)))  float           v8f;

__device__ __forceinline__ unsigned short f2bf(float f) {
  unsigned int u = __builtin_bit_cast(unsigned int, f);
  u += 0x7FFFu + ((u >> 16) & 1u);     // round-to-nearest-even
  return (unsigned short)(u >> 16);
}

// ---------------------------------------------------------------------------
// Fused linear: Y = act(bn(X @ W + b))   (bf16 WMMA, fp32 accumulate)
// X: [M, Cin] fp32 row-major, W: [Cin, Cout] fp32 row-major.
// flags: bit0 = batchnorm (g,beta), bit1 = relu
// Block = 256 threads = 8 waves; block tile 64x32; wave tile 16x16; K step 32.
//
// LDS tiles are stored in *fragment order*:
//  - A rows keep natural K order (lane half h takes k 16h..16h+15) so each lane's
//    16-element fragment is 32 contiguous bytes (2x ds_load_b128).
//  - B is stored transposed Bt[n][k] so a lane's fragment (fixed N column,
//    16 consecutive K) is 32 contiguous bytes (2x ds_load_b128).
// Row stride 48 ushorts (96B) keeps 16B chunks aligned and spreads banks.
// ---------------------------------------------------------------------------
#define TM 64
#define TN 32
#define TK 32
#define ASTR 48
#define BSTR 48

__global__ __launch_bounds__(256) void linear_wmma_kernel(
    const float* __restrict__ X, const float* __restrict__ W,
    const float* __restrict__ bias, const float* __restrict__ gg,
    const float* __restrict__ beta, float* __restrict__ Y,
    int M, int Cin, int Cout, int flags)
{
  __shared__ __align__(16) unsigned short AsU[64 * ASTR];
  __shared__ __align__(16) unsigned short BtU[32 * BSTR];

  const int tid  = threadIdx.x;
  const int lane = tid & 31;
  const int wid  = tid >> 5;        // 0..7
  const int wm   = wid & 3;         // wave row (M)
  const int wn   = wid >> 2;        // wave col (N)
  const int blockM = blockIdx.x * TM;
  const int blockN = blockIdx.y * TN;

  // A staging: row ar_ (0..63), K block acb in {0,8,16,24}
  const int ar_ = tid >> 2;
  const int acb = (tid & 3) << 3;
  // K-block -> slot-block identity {0,8,16,24} -> {0,8,16,24} (ushorts)
  const int agoff = acb;
  // B staging: K row bkr (0..31), N block bnb in {0,4,...,28}
  const int bkr = tid >> 3;
  const int bnb = (tid & 7) << 2;

  const bool fullMN = (blockM + TM <= M) && (blockN + TN <= Cout);

  v8f acc = {};

  for (int k0 = 0; k0 < Cin; k0 += TK) {
    if (fullMN && (k0 + TK <= Cin)) {
      // ---- fast path: vector global loads, packed LDS stores
      const float* xsrc = X + (size_t)(blockM + ar_) * Cin + k0 + acb;
      float2 f0 = *(const float2*)(xsrc + 0);
      float2 f1 = *(const float2*)(xsrc + 2);
      float2 f2 = *(const float2*)(xsrc + 4);
      float2 f3 = *(const float2*)(xsrc + 6);
      uint4 pk;
      pk.x = (unsigned)f2bf(f0.x) | ((unsigned)f2bf(f0.y) << 16);
      pk.y = (unsigned)f2bf(f1.x) | ((unsigned)f2bf(f1.y) << 16);
      pk.z = (unsigned)f2bf(f2.x) | ((unsigned)f2bf(f2.y) << 16);
      pk.w = (unsigned)f2bf(f3.x) | ((unsigned)f2bf(f3.y) << 16);
      *(uint4*)(&AsU[ar_ * ASTR + agoff]) = pk;

      const float* wsrc = W + (size_t)(k0 + bkr) * Cout + blockN + bnb;
      float4 w4 = *(const float4*)wsrc;
      BtU[(bnb + 0) * BSTR + bkr] = f2bf(w4.x);
      BtU[(bnb + 1) * BSTR + bkr] = f2bf(w4.y);
      BtU[(bnb + 2) * BSTR + bkr] = f2bf(w4.z);
      BtU[(bnb + 3) * BSTR + bkr] = f2bf(w4.w);
    } else {
      // ---- slow path: clamped index + post-select (no exec-divergent loads)
#pragma unroll
      for (int j = 0; j < 8; ++j) {
        int c = acb + j;
        int m = blockM + ar_, k = k0 + c;
        int mc = m < M   ? m : M - 1;
        int kc = k < Cin ? k : Cin - 1;
        float v = X[(size_t)mc * Cin + kc];
        v = (m < M && k < Cin) ? v : 0.0f;
        AsU[ar_ * ASTR + agoff + j] = f2bf(v);
      }
#pragma unroll
      for (int j = 0; j < 4; ++j) {
        int n = blockN + bnb + j;
        int k = k0 + bkr;
        int nc = n < Cout ? n : Cout - 1;
        int kc = k < Cin  ? k : Cin - 1;
        float v = W[(size_t)kc * Cout + nc];
        v = (n < Cout && k < Cin) ? v : 0.0f;
        BtU[(bnb + j) * BSTR + bkr] = f2bf(v);
      }
    }
    __syncthreads();

    // Fragment loads: 2x 16B contiguous LDS reads per operand.
    const unsigned short* ap =
        &AsU[((wm << 4) + (lane & 15)) * ASTR + ((lane >> 4) << 4)];
    uint4 a0 = *(const uint4*)(ap);
    uint4 a1 = *(const uint4*)(ap + 8);
    const unsigned short* bp =
        &BtU[((wn << 4) + (lane & 15)) * BSTR + ((lane >> 4) << 4)];
    uint4 b0 = *(const uint4*)(bp);
    uint4 b1 = *(const uint4*)(bp + 8);

    struct Pair { uint4 lo, hi; };
    Pair pa{a0, a1}, pb{b0, b1};
    v16bf a = __builtin_bit_cast(v16bf, pa);
    v16bf b = __builtin_bit_cast(v16bf, pb);
    acc = __builtin_amdgcn_wmma_f32_16x16x32_bf16(
        false, a, false, b, (short)0, acc, false, false);
    __syncthreads();
  }

  // Epilogue: bias + BN + ReLU fused, one HBM trip for the activation.
  const int ng    = blockN + (wn << 4) + (lane & 15);
  const int mbase = blockM + (wm << 4) + ((lane >> 4) << 3);
  if (ng < Cout) {
    float bsv = bias[ng];
    float gv  = (flags & 1) ? gg[ng]   : 1.0f;
    float bev = (flags & 1) ? beta[ng] : 0.0f;
#pragma unroll
    for (int r = 0; r < 8; ++r) {
      int m = mbase + r;
      if (m < M) {
        float v = acc[r] + bsv;
        if (flags & 1) v = v * gv + bev;
        if (flags & 2) v = v > 0.0f ? v : 0.0f;
        Y[(size_t)m * Cout + ng] = v;
      }
    }
  }
}

// ---------------------------------------------------------------------------
// Split xyz[:, :, 0:3] / xyz[:, :, 3:6]
// ---------------------------------------------------------------------------
__global__ void split_kernel(const float* __restrict__ xyz,
                             float* __restrict__ pts, float* __restrict__ nrm,
                             size_t npts)
{
  size_t t = (size_t)blockIdx.x * blockDim.x + threadIdx.x;
  if (t >= npts) return;
#pragma unroll
  for (int j = 0; j < 3; ++j) {
    pts[t * 3 + j] = xyz[t * 6 + j];
    nrm[t * 3 + j] = xyz[t * 6 + 3 + j];
  }
}

// ---------------------------------------------------------------------------
// Farthest point sampling: one workgroup per batch; running min-dist array in
// LDS (<= 4096 points -> 16 KB of the 320 KB WGP LDS).
// ---------------------------------------------------------------------------
__global__ __launch_bounds__(256) void fps_kernel(
    const float* __restrict__ pts, int n, int npoint, int* __restrict__ out)
{
  __shared__ float dists[4096];
  __shared__ float rv[256];
  __shared__ int   ri[256];
  __shared__ int   far_s;

  const int b = blockIdx.x, tid = threadIdx.x;
  const float* p = pts + (size_t)b * n * 3;

  for (int i = tid; i < n; i += 256) dists[i] = 1e10f;
  if (tid == 0) far_s = 0;
  __syncthreads();

  for (int it = 0; it < npoint; ++it) {
    const int far = far_s;
    if (tid == 0) out[b * npoint + it] = far;
    const float cx = p[far * 3], cy = p[far * 3 + 1], cz = p[far * 3 + 2];

    float bv = -1.0f; int bi = 0;
    for (int i = tid; i < n; i += 256) {
      float dx = p[i * 3] - cx, dy = p[i * 3 + 1] - cy, dz = p[i * 3 + 2] - cz;
      float d = dx * dx + dy * dy + dz * dz;
      float dm = dists[i];
      dm = d < dm ? d : dm;
      dists[i] = dm;
      if (dm > bv) { bv = dm; bi = i; }
    }
    rv[tid] = bv; ri[tid] = bi;
    __syncthreads();
    for (int s = 128; s > 0; s >>= 1) {
      if (tid < s) {
        if (rv[tid + s] > rv[tid] ||
            (rv[tid + s] == rv[tid] && ri[tid + s] < ri[tid])) {
          rv[tid] = rv[tid + s]; ri[tid] = ri[tid + s];
        }
      }
      __syncthreads();
    }
    if (tid == 0) far_s = ri[0];
    __syncthreads();
  }
}

// ---------------------------------------------------------------------------
// Gather rows: out[b,s,:] = src[b, idx[b,s], :]
// ---------------------------------------------------------------------------
__global__ void gather_kernel(const float* __restrict__ src,
                              const int* __restrict__ idx,
                              float* __restrict__ out,
                              int B_, int n, int S, int C)
{
  size_t t = (size_t)blockIdx.x * blockDim.x + threadIdx.x;
  size_t total = (size_t)B_ * S * C;
  if (t >= total) return;
  int c = (int)(t % C);
  size_t bs = t / C;
  int s = (int)(bs % S), b = (int)(bs / S);
  int id = idx[b * S + s];
  out[t] = src[((size_t)b * n + id) * C + c];
}

// ---------------------------------------------------------------------------
// Ball query: first `nsample` in-radius indices in ascending order, padded
// with the first in-radius index (0 if none) — matches reference semantics.
// ---------------------------------------------------------------------------
__global__ void ball_query_kernel(const float* __restrict__ xyz,
                                  const float* __restrict__ new_xyz,
                                  int B_, int n, int S, float r2, int K,
                                  int* __restrict__ gidx)
{
  int t = blockIdx.x * blockDim.x + threadIdx.x;
  if (t >= B_ * S) return;
  int b = t / S;
  const float* p = xyz + (size_t)b * n * 3;
  float qx = new_xyz[(size_t)t * 3];
  float qy = new_xyz[(size_t)t * 3 + 1];
  float qz = new_xyz[(size_t)t * 3 + 2];
  int* o = gidx + (size_t)t * K;
  int cnt = 0, first = 0;
  for (int i = 0; i < n && cnt < K; ++i) {
    float dx = p[i * 3] - qx, dy = p[i * 3 + 1] - qy, dz = p[i * 3 + 2] - qz;
    if (dx * dx + dy * dy + dz * dz <= r2) {
      if (cnt == 0) first = i;
      o[cnt++] = i;
    }
  }
  for (int k = cnt; k < K; ++k) o[k] = first;
}

// ---------------------------------------------------------------------------
// Build grouped features h[b,s,k,:] = [xyz[gidx]-new_xyz, norm[gidx], feat[gidx]]
// ---------------------------------------------------------------------------
__global__ void group_concat_kernel(const float* __restrict__ xyz,
                                    const float* __restrict__ nrm,
                                    const float* __restrict__ feat,
                                    const float* __restrict__ new_xyz,
                                    const int* __restrict__ gidx,
                                    float* __restrict__ h,
                                    int B_, int n, int S, int K, int Cf)
{
  const int Cin = 6 + Cf;
  size_t t = (size_t)blockIdx.x * blockDim.x + threadIdx.x;
  size_t total = (size_t)B_ * S * K * Cin;
  if (t >= total) return;
  int c = (int)(t % Cin);
  size_t r = t / Cin;
  int k = (int)(r % K);
  size_t bs = r / K;
  int b = (int)(bs / S);
  int id = gidx[bs * K + k];
  float v;
  if (c < 3)      v = xyz[((size_t)b * n + id) * 3 + c] - new_xyz[bs * 3 + c];
  else if (c < 6) v = nrm[((size_t)b * n + id) * 3 + (c - 3)];
  else            v = feat[((size_t)b * n + id) * Cf + (c - 6)];
  h[t] = v;
}

// ---------------------------------------------------------------------------
// Max pool over K neighbors: (rows, K, C) -> (rows, C)
// ---------------------------------------------------------------------------
__global__ void maxpool_kernel(const float* __restrict__ h,
                               float* __restrict__ out,
                               size_t rows, int K, int C)
{
  size_t t = (size_t)blockIdx.x * blockDim.x + threadIdx.x;
  size_t total = rows * (size_t)C;
  if (t >= total) return;
  int c = (int)(t % C);
  size_t r = t / C;
  float m = -1e30f;
  for (int k = 0; k < K; ++k) {
    float v = h[(r * K + k) * (size_t)C + c];
    m = v > m ? v : m;
  }
  out[t] = m;
}

// ---------------------------------------------------------------------------
// 3-NN search + inverse-distance weights
// ---------------------------------------------------------------------------
__global__ void three_nn_kernel(const float* __restrict__ x1,
                                const float* __restrict__ x2,
                                int B_, int S1, int S2,
                                int* __restrict__ idx3, float* __restrict__ w3)
{
  int t = blockIdx.x * blockDim.x + threadIdx.x;
  if (t >= B_ * S1) return;
  int b = t / S1;
  const float* p2 = x2 + (size_t)b * S2 * 3;
  float qx = x1[(size_t)t * 3], qy = x1[(size_t)t * 3 + 1], qz = x1[(size_t)t * 3 + 2];
  float d0 = 1e30f, d1 = 1e30f, d2 = 1e30f;
  int i0 = 0, i1 = 0, i2 = 0;
  for (int j = 0; j < S2; ++j) {
    float dx = p2[j * 3] - qx, dy = p2[j * 3 + 1] - qy, dz = p2[j * 3 + 2] - qz;
    float d = dx * dx + dy * dy + dz * dz;
    if (d < d0)      { d2 = d1; i2 = i1; d1 = d0; i1 = i0; d0 = d; i0 = j; }
    else if (d < d1) { d2 = d1; i2 = i1; d1 = d;  i1 = j; }
    else if (d < d2) { d2 = d;  i2 = j; }
  }
  float w0 = 1.0f / (d0 + 1e-8f);
  float w1 = 1.0f / (d1 + 1e-8f);
  float w2 = 1.0f / (d2 + 1e-8f);
  float s = w0 + w1 + w2;
  idx3[t * 3 + 0] = i0; idx3[t * 3 + 1] = i1; idx3[t * 3 + 2] = i2;
  w3[t * 3 + 0] = w0 / s; w3[t * 3 + 1] = w1 / s; w3[t * 3 + 2] = w2 / s;
}

// ---------------------------------------------------------------------------
// out[b,i,:] = concat(points1[b,i,:], sum_j w_j * points2[b, idx_j, :])
// ---------------------------------------------------------------------------
__global__ void interp_concat_kernel(const float* __restrict__ p1, int C1,
                                     const float* __restrict__ p2, int C2,
                                     const int* __restrict__ idx3,
                                     const float* __restrict__ w3,
                                     float* __restrict__ out,
                                     int B_, int S1, int S2)
{
  const int C = C1 + C2;
  size_t t = (size_t)blockIdx.x * blockDim.x + threadIdx.x;
  size_t total = (size_t)B_ * S1 * C;
  if (t >= total) return;
  int c = (int)(t % C);
  size_t row = t / C;
  int b = (int)(row / S1);
  if (c < C1) {
    out[t] = p1[row * (size_t)C1 + c];
  } else {
    int cc = c - C1;
    float acc = 0.0f;
#pragma unroll
    for (int j = 0; j < 3; ++j) {
      int id = idx3[row * 3 + j];
      acc += w3[row * 3 + j] * p2[((size_t)b * S2 + id) * (size_t)C2 + cc];
    }
    out[t] = acc;
  }
}

// ---------------------------------------------------------------------------
// log_softmax over 16 classes (one thread per row)
// ---------------------------------------------------------------------------
__global__ void log_softmax16_kernel(const float* __restrict__ in,
                                     float* __restrict__ out, size_t rows)
{
  size_t t = (size_t)blockIdx.x * blockDim.x + threadIdx.x;
  if (t >= rows) return;
  const float* x = in + t * 16;
  float m = x[0];
#pragma unroll
  for (int c = 1; c < 16; ++c) m = x[c] > m ? x[c] : m;
  float s = 0.0f;
#pragma unroll
  for (int c = 0; c < 16; ++c) s += __expf(x[c] - m);
  float ls = __logf(s);
#pragma unroll
  for (int c = 0; c < 16; ++c) out[t * 16 + c] = x[c] - m - ls;
}

__global__ void copy_kernel(const float* __restrict__ src,
                            float* __restrict__ dst, size_t n)
{
  size_t t = (size_t)blockIdx.x * blockDim.x + threadIdx.x;
  if (t < n) dst[t] = src[t];
}

// ---------------------------------------------------------------------------
// Host side
// ---------------------------------------------------------------------------
struct Lin { const float *W, *b, *g, *beta; int cin, cout; };

static inline dim3 blk1(size_t n) { return dim3((unsigned)((n + 255) / 256)); }

extern "C" void kernel_launch(void* const* d_in, const int* in_sizes, int n_in,
                              void* d_out, int out_size, void* d_ws, size_t ws_size,
                              hipStream_t stream)
{
  (void)in_sizes; (void)n_in; (void)out_size; (void)ws_size;
  const int Bc = 32, Nc = 4096;

  // -------- inputs (pytree leaf order: dicts sorted by key; lin = W,b,beta,g)
  const float* xyz = (const float*)d_in[0];
  int ci = 1;
  auto next_lin = [&](int cin, int cout) {
    Lin L;
    L.W    = (const float*)d_in[ci++];
    L.b    = (const float*)d_in[ci++];
    L.beta = (const float*)d_in[ci++];
    L.g    = (const float*)d_in[ci++];
    L.cin = cin; L.cout = cout;
    return L;
  };
  Lin conv1 = next_lin(128, 256);
  Lin conv2 = next_lin(256, 256);
  Lin fp0_0 = next_lin(128, 128);
  Lin fp1_0 = next_lin(320, 256), fp1_1 = next_lin(256, 128);
  Lin fp2_0 = next_lin(640, 512), fp2_1 = next_lin(512, 256);
  Lin fp3_0 = next_lin(768, 512), fp3_1 = next_lin(512, 512);
  Lin r1 = next_lin(128, 256), r2 = next_lin(256, 128);
  Lin r3 = next_lin(128, 64),  r4 = next_lin(64, 3);
  Lin sa0 = next_lin(6, 64),   sa1 = next_lin(70, 128);
  Lin sa2 = next_lin(134, 256), sa3 = next_lin(262, 512);
  Lin seg = next_lin(256, 16);

  // -------- workspace arena (all sizes multiples of 4 floats -> 16B aligned)
  char* base = (char*)d_ws;
  size_t off = 0;
  auto allocf = [&](size_t n) { float* p = (float*)(base + off); off += n * sizeof(float); return p; };
  auto alloci = [&](size_t n) { int*   p = (int*)  (base + off); off += n * sizeof(int);   return p; };

  float* pts  = allocf((size_t)Bc * Nc * 3);
  float* nrm  = allocf((size_t)Bc * Nc * 3);
  float* l0x  = allocf((size_t)Bc * 512 * 3); float* l0n = allocf((size_t)Bc * 512 * 3);
  float* l1x  = allocf((size_t)Bc * 256 * 3); float* l1n = allocf((size_t)Bc * 256 * 3);
  float* l2x  = allocf((size_t)Bc * 128 * 3); float* l2n = allocf((size_t)Bc * 128 * 3);
  float* l3x  = allocf((size_t)Bc * 64  * 3); float* l3n = allocf((size_t)Bc * 64  * 3);
  float* l0p  = allocf((size_t)Bc * 512 * 64);
  float* l1p  = allocf((size_t)Bc * 256 * 128);
  float* l2p  = allocf((size_t)Bc * 128 * 256);
  float* l3p  = allocf((size_t)Bc * 64  * 512);
  float* fp3o = allocf((size_t)Bc * 128 * 512);
  float* fp2o = allocf((size_t)Bc * 256 * 256);
  float* fp1o = allocf((size_t)Bc * 512 * 128);
  float* mid  = allocf((size_t)Bc * 256 * 512);   // FP intermediate
  int*   fidx = alloci((size_t)Bc * 512);
  int*   gidx = alloci((size_t)Bc * 512 * 8);     // 131072 for every SA level
  int*   idx3 = alloci((size_t)Bc * Nc * 3);
  float* w3   = allocf((size_t)Bc * Nc * 3);
  float* slabA = allocf((size_t)Bc * 64 * 64 * 262);  // 34.3M fl (grouped feats / reuse)
  float* slabB = allocf((size_t)Bc * 64 * 64 * 512);  // 67.1M fl (MLP out / reuse)

  float* segOut   = (float*)d_out;
  float* reconOut = segOut + (size_t)Bc * Nc * 16;
  float* l3pOut   = reconOut + (size_t)Bc * Nc * 3;

  auto linear = [&](const float* Xp, const Lin& L, float* Yp, int M, int flags) {
    dim3 g((M + TM - 1) / TM, (L.cout + TN - 1) / TN);
    linear_wmma_kernel<<<g, dim3(256), 0, stream>>>(
        Xp, L.W, L.b, L.g, L.beta, Yp, M, L.cin, L.cout, flags);
  };

  // -------- split xyz / normals
  split_kernel<<<blk1((size_t)Bc * Nc), 256, 0, stream>>>(xyz, pts, nrm, (size_t)Bc * Nc);

  // -------- set-abstraction helper
  auto run_sa = [&](const float* xl, const float* nl, const float* feat, int Cf,
                    int n, int S, int K, float r2, const Lin& L,
                    float* lx, float* ln, float* lp) {
    fps_kernel<<<dim3(Bc), dim3(256), 0, stream>>>(xl, n, S, fidx);
    gather_kernel<<<blk1((size_t)Bc * S * 3), 256, 0, stream>>>(xl, fidx, lx, Bc, n, S, 3);
    gather_kernel<<<blk1((size_t)Bc * S * 3), 256, 0, stream>>>(nl, fidx, ln, Bc, n, S, 3);
    ball_query_kernel<<<blk1((size_t)Bc * S), 256, 0, stream>>>(xl, lx, Bc, n, S, r2, K, gidx);
    size_t htot = (size_t)Bc * S * K * (6 + Cf);
    group_concat_kernel<<<blk1(htot), 256, 0, stream>>>(xl, nl, feat, lx, gidx,
                                                        slabA, Bc, n, S, K, Cf);
    linear(slabA, L, slabB, Bc * S * K, /*bn+relu*/3);
    maxpool_kernel<<<blk1((size_t)Bc * S * L.cout), 256, 0, stream>>>(
        slabB, lp, (size_t)Bc * S, K, L.cout);
  };

  run_sa(pts, nrm, nullptr, 0,   Nc,  512, 8,  0.04f, sa0, l0x, l0n, l0p);
  run_sa(l0x, l0n, l0p,     64,  512, 256, 16, 0.16f, sa1, l1x, l1n, l1p);
  run_sa(l1x, l1n, l1p,     128, 256, 128, 32, 0.36f, sa2, l2x, l2n, l2p);
  run_sa(l2x, l2n, l2p,     256, 128, 64,  64, 0.64f, sa3, l3x, l3n, l3p);

  // -------- feature propagation helper (3-NN interp + concat -> slabA)
  auto run_fp = [&](const float* x1, const float* x2, const float* p1, int C1,
                    const float* p2, int C2, int S1, int S2) {
    three_nn_kernel<<<blk1((size_t)Bc * S1), 256, 0, stream>>>(x1, x2, Bc, S1, S2, idx3, w3);
    interp_concat_kernel<<<blk1((size_t)Bc * S1 * (C1 + C2)), 256, 0, stream>>>(
        p1, C1, p2, C2, idx3, w3, slabA, Bc, S1, S2);
  };

  // FP3: (l2x <- l3x), 256+512 -> 512 -> 512
  run_fp(l2x, l3x, l2p, 256, l3p, 512, 128, 64);
  linear(slabA, fp3_0, mid,  Bc * 128, 3);
  linear(mid,   fp3_1, fp3o, Bc * 128, 3);
  // FP2: 128+512 -> 512 -> 256
  run_fp(l1x, l2x, l1p, 128, fp3o, 512, 256, 128);
  linear(slabA, fp2_0, mid,  Bc * 256, 3);
  linear(mid,   fp2_1, fp2o, Bc * 256, 3);
  // FP1: 64+256 -> 256 -> 128
  run_fp(l0x, l1x, l0p, 64, fp2o, 256, 512, 256);
  linear(slabA, fp1_0, mid,  Bc * 512, 3);
  linear(mid,   fp1_1, fp1o, Bc * 512, 3);
  // FP0: interp only (128) -> 128 ; f lives at slabB[0 : B*N*128]
  run_fp(pts, l0x, nullptr, 0, fp1o, 128, Nc, 512);
  float* f = slabB;
  linear(slabA, fp0_0, f, Bc * Nc, 3);

  // -------- heads
  float* feat1  = slabA;                                   // (B*N, 256)
  float* feat2  = slabB + (size_t)Bc * Nc * 128;           // (B*N, 256)
  float* seglin = feat2 + (size_t)Bc * Nc * 256;           // (B*N, 16)
  linear(f,     conv1, feat1,  Bc * Nc, 3);
  linear(feat1, conv2, feat2,  Bc * Nc, 3);
  linear(feat2, seg,   seglin, Bc * Nc, 0);
  log_softmax16_kernel<<<blk1((size_t)Bc * Nc), 256, 0, stream>>>(
      seglin, segOut, (size_t)Bc * Nc);

  // recon chain: f -> 256 -> 128 -> 64 -> 3
  float* rbufA = slabA;        // r1 out (feat1 dead)
  float* rbufB = slabB;        // r2 out (f dead after r1)
  linear(f,     r1, rbufA, Bc * Nc, 3);
  linear(rbufA, r2, rbufB, Bc * Nc, 3);
  linear(rbufB, r3, rbufA, Bc * Nc, 3);
  linear(rbufA, r4, reconOut, Bc * Nc, 0);

  // l3p output
  copy_kernel<<<blk1((size_t)Bc * 64 * 512), 256, 0, stream>>>(
      l3p, l3pOut, (size_t)Bc * 64 * 512);
}
